// RealPolarModelSimple_4887672783453
// MI455X (gfx1250) — matrix operand order repaired and checked by reading, not compiled, not run on hardware
//
#include <hip/hip_runtime.h>
#include <math.h>

typedef __attribute__((ext_vector_type(16))) _Float16 v16h;
typedef __attribute__((ext_vector_type(8)))  _Float16 v8h;
typedef __attribute__((ext_vector_type(4)))  _Float16 v4h;
typedef __attribute__((ext_vector_type(8)))  float    v8f;
typedef __attribute__((ext_vector_type(4)))  unsigned int u32x4;
typedef __attribute__((ext_vector_type(8)))  unsigned int u32x8;

#define B_      64
#define S_      1024
#define D_      256        // D_POLAR == D_INPUT == K
#define N2_     512        // 2*D_POLAR
#define M_      (B_ * S_)  // 65536 GEMM rows
#define MBLK_   512        // M_/128 row-blocks (8 per batch image)
#define KAN_    32
#define NCLS_   1000
#define NLAY_   8
#define PI_     3.14159265358979323846f
#define EPS_    1e-8f

__device__ __forceinline__ float gelu_exact(float v) {
    return 0.5f * v * (1.0f + erff(v * 0.70710678118654752f));
}

// ---------------------------------------------------------------------------
// K0a: transpose embed_w [K=256, N=512] -> Wt[n][k], split f16 hi + residual.
// ---------------------------------------------------------------------------
__global__ void k0_split_w(const float* __restrict__ W,
                           _Float16* __restrict__ WtHi,
                           _Float16* __restrict__ WtLo) {
    int idx = blockIdx.x * blockDim.x + threadIdx.x;   // 0 .. 131071
    int n = idx & (N2_ - 1);
    int k = idx >> 9;
    float f = W[(size_t)k * N2_ + n];
    _Float16 h = (_Float16)f;
    WtHi[(size_t)n * D_ + k] = h;
    WtLo[(size_t)n * D_ + k] = (_Float16)(f - (float)h);
}

// ---------------------------------------------------------------------------
// K0b: split x once into f16 hi + residual lo.
// ---------------------------------------------------------------------------
__global__ void k0_split_x(const float* __restrict__ x,
                           _Float16* __restrict__ XHi,
                           _Float16* __restrict__ XLo) {
    size_t idx = ((size_t)blockIdx.x * blockDim.x + threadIdx.x) * 4;
    float4 f = *(const float4*)(x + idx);
    v4h h, l;
    h[0] = (_Float16)f.x;  l[0] = (_Float16)(f.x - (float)h[0]);
    h[1] = (_Float16)f.y;  l[1] = (_Float16)(f.y - (float)h[1]);
    h[2] = (_Float16)f.z;  l[2] = (_Float16)(f.z - (float)h[2]);
    h[3] = (_Float16)f.w;  l[3] = (_Float16)(f.w - (float)h[3]);
    *(v4h*)(XHi + idx) = h;
    *(v4h*)(XLo + idx) = l;
}

// ---------------------------------------------------------------------------
// GEMM + polar pass (templated epilogue). Workgroup = 256 thr (8 waves),
// tile 128 rows x 32 d-cols (64 N-cols, pairing col d with col d+256).
// B panel DMA'd into LDS by the Tensor Data Mover (2-D descriptor, 2 blocks
// x 16KB, stride 128KB), wave 0 issues, TENSORcnt wait, ds_load fragments.
// FINAL==0: partial sums of (x0, y0) over the 128 rows -> Pa, Pb
// FINAL==1: partial sums of sqrt((x0+Cx)^2+(y0+Cy)^2+eps)  -> Pa
// Reduction is fixed-order (deterministic), LDS scratch reuses the B panel.
// ---------------------------------------------------------------------------
template <int FINAL>
__global__ void __launch_bounds__(256)
k_gemm_polar(const _Float16* __restrict__ XHi,
             const _Float16* __restrict__ XLo,
             const _Float16* __restrict__ WtHi,
             const _Float16* __restrict__ WtLo,
             const float* __restrict__ eb,
             const float* __restrict__ Cx, const float* __restrict__ Cy,
             float* __restrict__ Pa, float* __restrict__ Pb) {
    __shared__ _Float16 Bpanel[2][64 * 256];   // 64KB, dead after k-loop

    const int lane   = threadIdx.x & 31;
    const int wave   = threadIdx.x >> 5;
    const int bl16   = lane & 15;
    const bool hiLn  = lane >= 16;
    const int m0     = blockIdx.x * 128 + wave * 16;
    const int dBase  = blockIdx.y * 32;

    // ---- TDM: DMA the two B panels into LDS (wave 0 only) ----
    if (wave == 0) {
        const unsigned int ldsHi = (unsigned int)(uintptr_t)(&Bpanel[0][0]);
        const unsigned int ldsLo = (unsigned int)(uintptr_t)(&Bpanel[1][0]);
        const unsigned long long gaHi =
            (unsigned long long)(uintptr_t)(WtHi + (size_t)dBase * D_);
        const unsigned long long gaLo =
            (unsigned long long)(uintptr_t)(WtLo + (size_t)dBase * D_);
        // D# G1: data_size=8B, tensor=(2048,2), tile=(2048,2), stride0=16384
        const u32x8 g1 = { (3u << 16), (2048u << 16), (2u << 16),
                           (2048u << 16), 2u, 16384u, 0u, 0u };
        const u32x4 g0h = { 1u, ldsHi, (unsigned int)gaHi,
                           (unsigned int)((gaHi >> 32) & 0x01FFFFFFu) | 0x80000000u };
        const u32x4 g0l = { 1u, ldsLo, (unsigned int)gaLo,
                           (unsigned int)((gaLo >> 32) & 0x01FFFFFFu) | 0x80000000u };
        asm volatile("tensor_load_to_lds %0, %1" :: "s"(g0h), "s"(g1) : "memory");
        asm volatile("tensor_load_to_lds %0, %1" :: "s"(g0l), "s"(g1) : "memory");
        __builtin_amdgcn_s_wait_tensorcnt(0);
    }
    __syncthreads();

    v8f acc0 = {}, acc1 = {}, acc2 = {}, acc3 = {};

    const size_t rowOff = (size_t)(m0 + bl16) * D_;
    const _Float16* aHiP = XHi + rowOff;
    const _Float16* aLoP = XLo + rowOff;
    const int pc0 = bl16, pc1 = 16 + bl16, pc2 = 32 + bl16, pc3 = 48 + bl16;

    #pragma unroll
    for (int k0 = 0; k0 < D_; k0 += 32) {
        const int ka = k0 + (hiLn ? 16 : 0);
        v8h ah0 = *(const v8h*)(aHiP + ka);
        v8h ah1 = *(const v8h*)(aHiP + ka + 8);
        v8h al0 = *(const v8h*)(aLoP + ka);
        v8h al1 = *(const v8h*)(aLoP + ka + 8);
        v16h ah = __builtin_shufflevector(ah0, ah1, 0,1,2,3,4,5,6,7,8,9,10,11,12,13,14,15);
        v16h al = __builtin_shufflevector(al0, al1, 0,1,2,3,4,5,6,7,8,9,10,11,12,13,14,15);

        const int kb = k0 + (hiLn ? 16 : 0);
        {
            v16h bh = *(const v16h*)(&Bpanel[0][pc0 * D_ + kb]);
            v16h bl = *(const v16h*)(&Bpanel[1][pc0 * D_ + kb]);
            acc0 = __builtin_amdgcn_wmma_f32_16x16x32_f16(false, ah, false, bh, (short)0, acc0, false, false);
            acc0 = __builtin_amdgcn_wmma_f32_16x16x32_f16(false, ah, false, bl, (short)0, acc0, false, false);
            acc0 = __builtin_amdgcn_wmma_f32_16x16x32_f16(false, al, false, bh, (short)0, acc0, false, false);
        }
        {
            v16h bh = *(const v16h*)(&Bpanel[0][pc1 * D_ + kb]);
            v16h bl = *(const v16h*)(&Bpanel[1][pc1 * D_ + kb]);
            acc1 = __builtin_amdgcn_wmma_f32_16x16x32_f16(false, ah, false, bh, (short)0, acc1, false, false);
            acc1 = __builtin_amdgcn_wmma_f32_16x16x32_f16(false, ah, false, bl, (short)0, acc1, false, false);
            acc1 = __builtin_amdgcn_wmma_f32_16x16x32_f16(false, al, false, bh, (short)0, acc1, false, false);
        }
        {
            v16h bh = *(const v16h*)(&Bpanel[0][pc2 * D_ + kb]);
            v16h bl = *(const v16h*)(&Bpanel[1][pc2 * D_ + kb]);
            acc2 = __builtin_amdgcn_wmma_f32_16x16x32_f16(false, ah, false, bh, (short)0, acc2, false, false);
            acc2 = __builtin_amdgcn_wmma_f32_16x16x32_f16(false, ah, false, bl, (short)0, acc2, false, false);
            acc2 = __builtin_amdgcn_wmma_f32_16x16x32_f16(false, al, false, bh, (short)0, acc2, false, false);
        }
        {
            v16h bh = *(const v16h*)(&Bpanel[0][pc3 * D_ + kb]);
            v16h bl = *(const v16h*)(&Bpanel[1][pc3 * D_ + kb]);
            acc3 = __builtin_amdgcn_wmma_f32_16x16x32_f16(false, ah, false, bh, (short)0, acc3, false, false);
            acc3 = __builtin_amdgcn_wmma_f32_16x16x32_f16(false, ah, false, bl, (short)0, acc3, false, false);
            acc3 = __builtin_amdgcn_wmma_f32_16x16x32_f16(false, al, false, bh, (short)0, acc3, false, false);
        }
    }

    // ---- epilogue: bias, polar -> cartesian, per-lane row sums ----
    const int bb = blockIdx.x >> 3;            // batch index (8 row-blocks/b)
    float sA[2] = {0.f, 0.f};                  // FINAL: sum r ; else: sum x0
    float sB[2] = {0.f, 0.f};                  // sum y0 (means pass only)
    #pragma unroll
    for (int t = 0; t < 2; ++t) {
        const v8f aLo = (t == 0) ? acc0 : acc1;
        const v8f aHi = (t == 0) ? acc2 : acc3;
        const int dcol = dBase + t * 16 + bl16;
        const float bLo = eb[dcol];
        const float bHi = eb[D_ + dcol];
        float cxv = 0.f, cyv = 0.f;
        if constexpr (FINAL) {
            cxv = Cx[bb * D_ + dcol];
            cyv = Cy[bb * D_ + dcol];
        }
        #pragma unroll
        for (int j = 0; j < 8; ++j) {
            float hlo = aLo[j] + bLo;
            float hhi = aHi[j] + bHi;
            float r   = fabsf(hlo) + 0.1f;
            float th  = hhi * PI_;
            float sn, cs;
            sincosf(th, &sn, &cs);
            float x0 = r * cs, y0 = r * sn;
            if constexpr (FINAL) {
                float xx = x0 + cxv, yy = y0 + cyv;
                sA[t] += sqrtf(xx * xx + yy * yy + EPS_);
            } else {
                sA[t] += x0;
                sB[t] += y0;
            }
        }
    }

    // ---- deterministic workgroup reduction (reuse B panel LDS) ----
    __syncthreads();                            // all waves done reading B
    float* red = (float*)&Bpanel[0][0];
    red[(wave * 2 + 0) * 32 + lane] = sA[0];
    red[(wave * 2 + 1) * 32 + lane] = sA[1];
    if constexpr (!FINAL) {
        red[512 + (wave * 2 + 0) * 32 + lane] = sB[0];
        red[512 + (wave * 2 + 1) * 32 + lane] = sB[1];
    }
    __syncthreads();
    if (threadIdx.x < (FINAL ? 32u : 64u)) {
        const int c = threadIdx.x & 31;         // column within the 32-d tile
        const float* base = red + ((threadIdx.x >= 32) ? 512 : 0);
        const int t = c >> 4, cl = c & 15;
        float s = 0.f;
        #pragma unroll
        for (int w = 0; w < 8; ++w) {           // fixed order: deterministic
            const float* rp = base + (w * 2 + t) * 32;
            s += rp[cl] + rp[cl + 16];
        }
        float* dst = (threadIdx.x >= 32) ? Pb : Pa;
        dst[(size_t)blockIdx.x * D_ + dBase + c] = s;
    }
}

// ---------------------------------------------------------------------------
// Reduce the 8 row-block partials of one batch image -> mean over S.
// ---------------------------------------------------------------------------
__global__ void k_reduce8(const float* __restrict__ P, float* __restrict__ out) {
    const int idx = blockIdx.x * blockDim.x + threadIdx.x;   // 0..16383
    const int b = idx >> 8, d = idx & (D_ - 1);
    const float* p = P + (size_t)b * 8 * D_ + d;
    float s = 0.f;
    #pragma unroll
    for (int i = 0; i < 8; ++i) s += p[i * D_];
    out[idx] = s * (1.0f / S_);
}

// ---------------------------------------------------------------------------
// K3: 8-layer loop on the [B,D] mean state; tiny KAN MLPs; trig-free.
// ---------------------------------------------------------------------------
__global__ void k3_layers(const float* __restrict__ xm0, const float* __restrict__ ym0,
                          const float* __restrict__ mw1, const float* __restrict__ mb1,
                          const float* __restrict__ mw2, const float* __restrict__ mb2,
                          const float* __restrict__ pw1, const float* __restrict__ pb1,
                          const float* __restrict__ pw2, const float* __restrict__ pb2,
                          float* __restrict__ Cx, float* __restrict__ Cy) {
    const int idx = blockIdx.x * blockDim.x + threadIdx.x;   // 0..16383
    float xm = xm0[idx], ym = ym0[idx];
    float cxs = 0.f, cys = 0.f;
    for (int i = 0; i < NLAY_; ++i) {
        float ra  = sqrtf(xm * xm + ym * ym + EPS_);     // r_agg
        float lr  = logf(ra + EPS_);
        const float* w1 = mw1 + i * KAN_;
        const float* b1 = mb1 + i * KAN_;
        const float* w2 = mw2 + i * KAN_;
        float acc = mb2[i];
        #pragma unroll 8
        for (int j = 0; j < KAN_; ++j)
            acc += gelu_exact(lr * w1[j] + b1[j]) * w2[j];
        float rt  = expf(acc);                           // r_trans
        float inv = 1.0f / ra;
        float sn = ym * inv, cs = xm * inv;              // sin/cos(th_agg)
        const float* q1 = pw1 + i * 2 * KAN_;            // [2][32]
        const float* qb = pb1 + i * KAN_;
        const float* q2 = pw2 + i * KAN_ * 2;            // [32][2]
        float p0 = pb2[i * 2 + 0], p1 = pb2[i * 2 + 1];
        #pragma unroll 8
        for (int j = 0; j < KAN_; ++j) {
            float g = gelu_exact(sn * q1[j] + cs * q1[KAN_ + j] + qb[j]);
            p0 += g * q2[j * 2 + 0];
            p1 += g * q2[j * 2 + 1];
        }
        float nrm = sqrtf(p0 * p0 + p1 * p1 + EPS_);
        float cx = rt * (p1 / nrm);                      // rt*cos(th_trans)
        float cy = rt * (p0 / nrm);                      // rt*sin(th_trans)
        xm += cx;  ym += cy;
        cxs += cx; cys += cy;
    }
    Cx[idx] = cxs;
    Cy[idx] = cys;
}

// ---------------------------------------------------------------------------
// K5a/K5b: classifier 256 -> 32 (gelu) -> 1000  (2.6 MFLOP, plain FMA)
// ---------------------------------------------------------------------------
__global__ void k5a_hidden(const float* __restrict__ pooled,
                           const float* __restrict__ w1, const float* __restrict__ b1,
                           float* __restrict__ hidden) {
    const int idx = blockIdx.x * blockDim.x + threadIdx.x;   // 0..2047
    const int m = idx >> 5, j = idx & 31;
    const float* p = pooled + m * D_;
    float acc = b1[j];
    #pragma unroll 8
    for (int k = 0; k < D_; ++k)
        acc += p[k] * w1[k * KAN_ + j];
    hidden[idx] = gelu_exact(acc);
}

__global__ void k5b_out(const float* __restrict__ hidden,
                        const float* __restrict__ w2, const float* __restrict__ b2,
                        float* __restrict__ out) {
    const int idx = blockIdx.x * blockDim.x + threadIdx.x;   // 0..63999
    if (idx >= B_ * NCLS_) return;
    const int m = idx / NCLS_, n = idx - m * NCLS_;
    const float* h = hidden + m * KAN_;
    float acc = b2[n];
    #pragma unroll
    for (int j = 0; j < KAN_; ++j)
        acc += h[j] * w2[j * NCLS_ + n];
    out[idx] = acc;
}

// ---------------------------------------------------------------------------
extern "C" void kernel_launch(void* const* d_in, const int* in_sizes, int n_in,
                              void* d_out, int out_size, void* d_ws, size_t ws_size,
                              hipStream_t stream) {
    const float* x    = (const float*)d_in[0];
    const float* ew   = (const float*)d_in[1];
    const float* eb   = (const float*)d_in[2];
    const float* mw1  = (const float*)d_in[3];
    const float* mb1  = (const float*)d_in[4];
    const float* mw2  = (const float*)d_in[5];
    const float* mb2  = (const float*)d_in[6];
    const float* pw1  = (const float*)d_in[7];
    const float* pb1  = (const float*)d_in[8];
    const float* pw2  = (const float*)d_in[9];
    const float* pb2  = (const float*)d_in[10];
    const float* cw1  = (const float*)d_in[11];
    const float* cb1  = (const float*)d_in[12];
    const float* cw2  = (const float*)d_in[13];
    const float* cb2  = (const float*)d_in[14];
    float* out = (float*)d_out;

    char* ws = (char*)d_ws;
    size_t off = 0;
    _Float16* WtHi = (_Float16*)(ws + off); off += (size_t)N2_ * D_ * 2;    // 256KB
    _Float16* WtLo = (_Float16*)(ws + off); off += (size_t)N2_ * D_ * 2;    // 256KB
    _Float16* XHi  = (_Float16*)(ws + off); off += (size_t)M_ * D_ * 2;     // 32MB
    _Float16* XLo  = (_Float16*)(ws + off); off += (size_t)M_ * D_ * 2;     // 32MB
    float* Sx      = (float*)(ws + off);    off += (size_t)MBLK_ * D_ * 4;  // 512KB
    float* Sy      = (float*)(ws + off);    off += (size_t)MBLK_ * D_ * 4;  // 512KB
    float* Sr      = (float*)(ws + off);    off += (size_t)MBLK_ * D_ * 4;  // 512KB
    float* xm0     = (float*)(ws + off);    off += (size_t)B_ * D_ * 4;
    float* ym0     = (float*)(ws + off);    off += (size_t)B_ * D_ * 4;
    float* Cx      = (float*)(ws + off);    off += (size_t)B_ * D_ * 4;
    float* Cy      = (float*)(ws + off);    off += (size_t)B_ * D_ * 4;
    float* pooled  = (float*)(ws + off);    off += (size_t)B_ * D_ * 4;
    float* hidden  = (float*)(ws + off);    off += (size_t)B_ * KAN_ * 4;

    k0_split_w<<<(N2_ * D_) / 256, 256, 0, stream>>>(ew, WtHi, WtLo);
    k0_split_x<<<(M_ * D_ / 4) / 256, 256, 0, stream>>>(x, XHi, XLo);

    // pass 1: GEMM + polar + mean partials (no XY materialization)
    k_gemm_polar<0><<<dim3(MBLK_, D_ / 32), 256, 0, stream>>>(
        XHi, XLo, WtHi, WtLo, eb, nullptr, nullptr, Sx, Sy);
    k_reduce8<<<(B_ * D_) / 256, 256, 0, stream>>>(Sx, xm0);
    k_reduce8<<<(B_ * D_) / 256, 256, 0, stream>>>(Sy, ym0);

    k3_layers<<<(B_ * D_) / 256, 256, 0, stream>>>(xm0, ym0, mw1, mb1, mw2, mb2,
                                                   pw1, pb1, pw2, pb2, Cx, Cy);

    // pass 2: recompute GEMM, shift by (Cx,Cy), pooled-r partials
    k_gemm_polar<1><<<dim3(MBLK_, D_ / 32), 256, 0, stream>>>(
        XHi, XLo, WtHi, WtLo, eb, Cx, Cy, Sr, Sr);
    k_reduce8<<<(B_ * D_) / 256, 256, 0, stream>>>(Sr, pooled);

    k5a_hidden<<<(B_ * KAN_) / 256, 256, 0, stream>>>(pooled, cw1, cb1, hidden);
    k5b_out<<<(B_ * NCLS_ + 255) / 256, 256, 0, stream>>>(hidden, cw2, cb2, out);
}
